// Local_Aggregation_89721866813964
// MI455X (gfx1250) — matrix operand order repaired and checked by reading, not compiled, not run on hardware
//
#include <hip/hip_runtime.h>
#include <math.h>

typedef _Float16 f16;
typedef __attribute__((ext_vector_type(16))) _Float16 v16h;
typedef __attribute__((ext_vector_type(8)))  _Float16 v8h;
typedef __attribute__((ext_vector_type(8)))  float    v8f;
typedef __attribute__((ext_vector_type(4)))  float    v4f;
typedef __attribute__((ext_vector_type(2)))  float    v2f;

#define C32   32
#define HW    4096      // 64*64
#define NPAIR 144
#define NVIEW 25
#define OFFS_STRIDE 24  // 18 offset channels padded to 24 floats per pixel

// ---- neighbor tables (from _build_nbrs in the reference) -------------------
__constant__ int SVI[NPAIR] = {
  1,5,6,
  0,2,5,6,7,
  1,3,6,7,8,
  2,4,7,8,9,
  3,8,9,
  0,1,6,10,11,
  0,1,2,5,7,10,11,12,
  1,2,3,6,8,11,12,13,
  2,3,4,7,9,12,13,14,
  3,4,8,13,14,
  5,6,11,15,16,
  5,6,7,10,12,15,16,17,
  6,7,8,11,13,16,17,18,
  7,8,9,12,14,17,18,19,
  8,9,13,18,19,
  10,11,16,20,21,
  10,11,12,15,17,20,21,22,
  11,12,13,16,18,21,22,23,
  12,13,14,17,19,22,23,24,
  13,14,18,23,24,
  15,16,21,
  15,16,17,20,22,
  16,17,18,21,23,
  17,18,19,22,24,
  18,19,23
};
__constant__ int CVI[NPAIR] = {
  0,0,0,
  1,1,1,1,1,
  2,2,2,2,2,
  3,3,3,3,3,
  4,4,4,
  5,5,5,5,5,
  6,6,6,6,6,6,6,6,
  7,7,7,7,7,7,7,7,
  8,8,8,8,8,8,8,8,
  9,9,9,9,9,
  10,10,10,10,10,
  11,11,11,11,11,11,11,11,
  12,12,12,12,12,12,12,12,
  13,13,13,13,13,13,13,13,
  14,14,14,14,14,
  15,15,15,15,15,
  16,16,16,16,16,16,16,16,
  17,17,17,17,17,17,17,17,
  18,18,18,18,18,18,18,18,
  19,19,19,19,19,
  20,20,20,
  21,21,21,21,21,
  22,22,22,22,22,
  23,23,23,23,23,
  24,24,24
};
__constant__ int STARTS[NVIEW] = {0,3,8,13,18,21,26,34,42,50,55,60,68,76,84,
                                  89,94,102,110,118,123,126,131,136,141};
__constant__ int NCNT[NVIEW]   = {3,5,5,5,3, 5,8,8,8,5, 5,8,8,8,5,
                                  5,8,8,8,5, 3,5,5,5,3};

// ---- WMMA helpers ----------------------------------------------------------
__device__ __forceinline__ int laneid() { return threadIdx.x & 31; }

__device__ __forceinline__ v8f wmma32(v16h a, v16h b, v8f c) {
  return __builtin_amdgcn_wmma_f32_16x16x32_f16(false, a, false, b,
                                                (short)0, c, false, false);
}

__device__ __forceinline__ v16h zero16() { v16h z = {}; return z; }

// B fragment (32x16): 16 pixels from pix0, 32 channels, NHWC f16 image.
__device__ __forceinline__ v16h bfrag(const f16* img, int pix0) {
  int l = laneid();
  return *(const v16h*)(img + (size_t)(pix0 + (l & 15)) * C32 + ((l >> 4) << 4));
}

// B fragment with zero padding for shifted 3x3 taps.
__device__ __forceinline__ v16h bfrag_pad(const f16* img, int y, int x) {
  int l = laneid();
  int xx = x + (l & 15);
  if ((unsigned)y < 64u && (unsigned)xx < 64u)
    return *(const v16h*)(img + (size_t)(y * 64 + xx) * C32 + ((l >> 4) << 4));
  return zero16();
}

// A fragment (16x32) from row-major f16 weights W[32][Krow] (contiguous k = 16h + s, matches B frags).
// Works for global or LDS pointers (LDS -> ds_load_b128 x2 after inlining).
__device__ __forceinline__ v16h afrag(const f16* W, int Krow, int mt, int kc) {
  int l = laneid();
  const f16* row = W + (size_t)(mt * 16 + (l & 15)) * Krow + kc + ((l >> 4) << 4);
  union { v16h v; v8h h[2]; } u;
  u.h[0] = *(const v8h*)(row);
  u.h[1] = *(const v8h*)(row + 8);
  return u.v;
}

// Stage a weight matrix (nhalf f16 elements, multiple of 8) into LDS.
__device__ __forceinline__ void stage_lds(f16* dst, const f16* src, int nhalf) {
  v8h* d = (v8h*)dst;
  const v8h* s = (const v8h*)src;
  for (int i = threadIdx.x; i < (nhalf >> 3); i += blockDim.x) d[i] = s[i];
  __syncthreads();
}

// Preload the 8 per-lane bias values for M-tile mt as one 32B load.
__device__ __forceinline__ v8f load_bias(const float* bias, int mt) {
  int l = laneid();
  return *(const v8f*)(bias + mt * 16 + ((l >> 4) << 3));
}

// C/D store to NHWC f16: bias + optional leaky-relu, single 16B store.
template <bool ACT>
__device__ __forceinline__ void store_nhwc(f16* out, int pix0, int mt, v8f c, v8f bias) {
  int l = laneid();
  int mo = (l >> 4) << 3;
  v8h h;
  for (int j = 0; j < 8; ++j) {
    float v = c[j] + bias[j];
    if (ACT) v = v >= 0.0f ? v : 0.1f * v;
    h[j] = (f16)v;
  }
  *(v8h*)(out + (size_t)(pix0 + (l & 15)) * C32 + mt * 16 + mo) = h;
}

// ---- weight / activation repack kernels ------------------------------------
__global__ void k_cvt_direct(const float* __restrict__ s, f16* __restrict__ d, int n) {
  int i = blockIdx.x * blockDim.x + threadIdx.x;
  if (i < n) d[i] = (f16)s[i];
}

// OIHW (32,Ci,3,3) fp32 -> f16 [32][ tap*Ci + ci ], tap = ky*3+kx
__global__ void k_cvt_oihw(const float* __restrict__ s, f16* __restrict__ d, int Ci) {
  int i = blockIdx.x * blockDim.x + threadIdx.x;
  int n = 32 * Ci * 9;
  if (i >= n) return;
  int m  = i / (9 * Ci);
  int r  = i - m * 9 * Ci;
  int tap = r / Ci, ci = r - tap * Ci;
  int ky = tap / 3, kx = tap - ky * 3;
  d[i] = (f16)s[((m * Ci + ci) * 3 + ky) * 3 + kx];
}

// w_off (18,32,1,1) -> f16 [32][32], rows 18..31 zero-padded
__global__ void k_cvt_off(const float* __restrict__ s, f16* __restrict__ d) {
  int i = blockIdx.x * blockDim.x + threadIdx.x;
  if (i >= 32 * 32) return;
  int m = i >> 5;
  d[i] = (m < 18) ? (f16)s[i] : (f16)0.0f;
}

// feature fp32 NCHW [25][32][4096] -> f16 NHWC [25][4096][32]
__global__ void k_feat_nhwc(const float* __restrict__ s, f16* __restrict__ d) {
  int i = blockIdx.x * blockDim.x + threadIdx.x;
  if (i >= NVIEW * HW * C32) return;
  int ch  = i & 31;
  int pix = (i >> 5) & (HW - 1);
  int v   = i / (HW * C32);
  d[i] = (f16)s[(size_t)(v * C32 + ch) * HW + pix];
}

// ---- stage kernels ---------------------------------------------------------
// conv1: buf = lrelu(W1 * [sv; cv] + b). Dual-tile; 4 A-frags live in VGPRs.
__global__ void k_conv1(const f16* __restrict__ feat, const f16* __restrict__ W1,
                        const float* __restrict__ b1, f16* __restrict__ buf) {
  int pair = blockIdx.x;
  const f16* sv = feat + (size_t)SVI[pair] * HW * C32;
  const f16* cv = feat + (size_t)CVI[pair] * HW * C32;
  f16* out = buf + (size_t)pair * HW * C32;
  int wave = threadIdx.x >> 5;
  v16h a00 = afrag(W1, 64, 0, 0),  a01 = afrag(W1, 64, 0, 32);
  v16h a10 = afrag(W1, 64, 1, 0),  a11 = afrag(W1, 64, 1, 32);
  v8f bb0 = load_bias(b1, 0), bb1 = load_bias(b1, 1);
  int tbase = blockIdx.y * 64 + wave * 8;
  for (int it = 0; it < 8; it += 2) {
    int p0 = (tbase + it) * 16, p1 = p0 + 16;
    v16h bs0 = bfrag(sv, p0), bc0 = bfrag(cv, p0);
    v16h bs1 = bfrag(sv, p1), bc1 = bfrag(cv, p1);
    v8f c00 = {}, c01 = {}, c10 = {}, c11 = {};
    c00 = wmma32(a00, bs0, c00); c00 = wmma32(a01, bc0, c00);
    c01 = wmma32(a00, bs1, c01); c01 = wmma32(a01, bc1, c01);
    c10 = wmma32(a10, bs0, c10); c10 = wmma32(a11, bc0, c10);
    c11 = wmma32(a10, bs1, c11); c11 = wmma32(a11, bc1, c11);
    store_nhwc<true>(out, p0, 0, c00, bb0);
    store_nhwc<true>(out, p1, 0, c01, bb0);
    store_nhwc<true>(out, p0, 1, c10, bb1);
    store_nhwc<true>(out, p1, 1, c11, bb1);
  }
}

// ASPP branch: b_br = lrelu(conv3x3(buf, Wa_br, dil=1<<br)).
// Weights staged in LDS; B stream stays on the vector-memory port.
__global__ void k_aspp_branch(const f16* __restrict__ buf, const f16* __restrict__ Wa,
                              f16* __restrict__ b3) {
  __shared__ f16 sW[32 * 288];
  int branch = blockIdx.x % 3;
  int pair   = blockIdx.x / 3;
  int dil = 1 << branch;
  stage_lds(sW, Wa + (size_t)branch * 32 * 288, 32 * 288);
  const f16* in = buf + (size_t)pair * HW * C32;
  f16* out = b3 + ((size_t)pair * 3 + branch) * HW * C32;
  int wave = threadIdx.x >> 5;
  v8f zb = {};
  int tbase = blockIdx.y * 64 + wave * 8;
  for (int it = 0; it < 8; it += 2) {
    int t0 = tbase + it;                 // even -> t0,t0+1 share a row
    int y = t0 >> 2, x0 = (t0 & 3) << 4;
    v8f c00 = {}, c01 = {}, c10 = {}, c11 = {};
    for (int tap = 0; tap < 9; ++tap) {
      int ky = tap / 3, kx = tap - ky * 3;
      int yy = y + dil * (ky - 1), xs = x0 + dil * (kx - 1);
      v16h b0 = bfrag_pad(in, yy, xs);
      v16h b1 = bfrag_pad(in, yy, xs + 16);
      v16h a0 = afrag(sW, 288, 0, tap * 32);
      v16h a1 = afrag(sW, 288, 1, tap * 32);
      c00 = wmma32(a0, b0, c00); c01 = wmma32(a0, b1, c01);
      c10 = wmma32(a1, b0, c10); c11 = wmma32(a1, b1, c11);
    }
    store_nhwc<true>(out, t0 * 16,      0, c00, zb);
    store_nhwc<true>(out, t0 * 16 + 16, 0, c01, zb);
    store_nhwc<true>(out, t0 * 16,      1, c10, zb);
    store_nhwc<true>(out, t0 * 16 + 16, 1, c11, zb);
  }
}

// ASPP transform: buf += conv3x3([b1;b2;b3], Wat). Wat (54KB) staged in LDS.
__global__ void k_aspp_fuse(const f16* __restrict__ b3, const f16* __restrict__ Wat,
                            f16* __restrict__ buf) {
  __shared__ f16 sW[32 * 864];
  int pair = blockIdx.x;
  stage_lds(sW, Wat, 32 * 864);
  const f16* bin = b3 + (size_t)pair * 3 * HW * C32;
  f16* bio = buf + (size_t)pair * HW * C32;
  int wave = threadIdx.x >> 5;
  int l = laneid();
  int mo = (l >> 4) << 3;
  int tbase = blockIdx.y * 64 + wave * 8;
  for (int it = 0; it < 8; it += 2) {
    int t0 = tbase + it;
    int y = t0 >> 2, x0 = (t0 & 3) << 4;
    v8f c00 = {}, c01 = {}, c10 = {}, c11 = {};
    for (int tap = 0; tap < 9; ++tap) {
      int ky = tap / 3, kx = tap - ky * 3;
      int yy = y + ky - 1, xs = x0 + kx - 1;
      for (int br = 0; br < 3; ++br) {
        const f16* src = bin + (size_t)br * HW * C32;
        v16h b0 = bfrag_pad(src, yy, xs);
        v16h b1 = bfrag_pad(src, yy, xs + 16);
        int kc = tap * 96 + br * 32;
        v16h a0 = afrag(sW, 864, 0, kc);
        v16h a1 = afrag(sW, 864, 1, kc);
        c00 = wmma32(a0, b0, c00); c01 = wmma32(a0, b1, c01);
        c10 = wmma32(a1, b0, c10); c11 = wmma32(a1, b1, c11);
      }
    }
    for (int s = 0; s < 2; ++s) {
      v8f cl = s ? c01 : c00, ch = s ? c11 : c10;
      v8h* p = (v8h*)(bio + (size_t)(t0 * 16 + s * 16 + (l & 15)) * C32 + mo);
      v8h o0 = p[0], o1 = p[2];
      v8h r0, r1;
      for (int j = 0; j < 8; ++j) r0[j] = (f16)((float)o0[j] + cl[j]);
      for (int j = 0; j < 8; ++j) r1[j] = (f16)((float)o1[j] + ch[j]);
      p[0] = r0; p[2] = r1;
    }
  }
}

// offset conv 1x1 (32 -> 18 padded). Output pixel-major [pair][pix][24] fp32.
__global__ void k_offset(const f16* __restrict__ buf, const f16* __restrict__ Woff,
                         const float* __restrict__ boff, float* __restrict__ offs) {
  int pair = blockIdx.x;
  const f16* in = buf + (size_t)pair * HW * C32;
  float* out = offs + (size_t)pair * HW * OFFS_STRIDE;
  int wave = threadIdx.x >> 5;
  int l = laneid();
  int mo = (l >> 4) << 3;
  v16h a0 = afrag(Woff, 32, 0, 0);
  v16h a1 = afrag(Woff, 32, 1, 0);
  v8f bb0 = *(const v8f*)(boff + mo);    // channels mo..mo+7 (all < 16, in range)
  float b16 = boff[16], b17 = boff[17];
  int tbase = blockIdx.y * 64 + wave * 8;
  for (int it = 0; it < 8; ++it) {
    int pix0 = (tbase + it) * 16;
    v16h b = bfrag(in, pix0);
    v8f c0 = {}, c1 = {};
    c0 = wmma32(a0, b, c0);
    c1 = wmma32(a1, b, c1);
    int pix = pix0 + (l & 15);
    float* q = out + (size_t)pix * OFFS_STRIDE;
    v4f f0 = {c0[0] + bb0[0], c0[1] + bb0[1], c0[2] + bb0[2], c0[3] + bb0[3]};
    v4f f1 = {c0[4] + bb0[4], c0[5] + bb0[5], c0[6] + bb0[6], c0[7] + bb0[7]};
    *(v4f*)(q + mo)     = f0;
    *(v4f*)(q + mo + 4) = f1;
    if (mo == 0) { q[16] = c1[0] + b16; q[17] = c1[1] + b17; }
  }
}

__device__ __forceinline__ v16h texld(const f16* img, int y, int x, int kg) {
  if ((unsigned)y < 64u && (unsigned)x < 64u)
    return *(const v16h*)(img + (size_t)(y * 64 + x) * C32 + kg);
  return zero16();
}

// deformable conv: packed-f16 bilinear blend feeds WMMA B-frags; Wd in LDS.
__global__ void k_dcn(const f16* __restrict__ feat, const float* __restrict__ offs,
                      const f16* __restrict__ Wd, const float* __restrict__ bd,
                      f16* __restrict__ aligned) {
  __shared__ f16 sW[32 * 288];
  int pair = blockIdx.x;
  stage_lds(sW, Wd, 32 * 288);
  const f16* img = feat + (size_t)SVI[pair] * HW * C32;
  const float* of = offs + (size_t)pair * HW * OFFS_STRIDE;
  f16* out = aligned + (size_t)pair * HW * C32;
  int wave = threadIdx.x >> 5;
  int l = laneid();
  int kg = (l >> 4) << 4;
  v8f bb0 = load_bias(bd, 0), bb1 = load_bias(bd, 1);
  int tbase = blockIdx.y * 64 + wave * 8;
  for (int it = 0; it < 8; ++it) {
    int t = tbase + it;
    int y = t >> 2;
    int x = ((t & 3) << 4) + (l & 15);
    int pix = y * 64 + x;
    const float* op = of + (size_t)pix * OFFS_STRIDE;
    v8f c0 = {}, c1 = {};
    for (int tap = 0; tap < 9; ++tap) {
      int ki = tap / 3, kj = tap - ki * 3;
      v2f d = *(const v2f*)(op + tap * 2);   // (dy, dx) in one b64
      float py = (float)y + (float)(ki - 1) + d.x;
      float px = (float)x + (float)(kj - 1) + d.y;
      float fy = floorf(py), fx = floorf(px);
      int y0 = (int)fy, x0 = (int)fx;
      float wy = py - fy, wx = px - fx;
      v16h v00 = texld(img, y0,     x0,     kg);
      v16h v01 = texld(img, y0,     x0 + 1, kg);
      v16h v10 = texld(img, y0 + 1, x0,     kg);
      v16h v11 = texld(img, y0 + 1, x0 + 1, kg);
      // packed f16 blend (v_pk_fma_f16): error same order as f16 quantization
      f16 h00 = (f16)((1.f - wy) * (1.f - wx)), h01 = (f16)((1.f - wy) * wx);
      f16 h10 = (f16)(wy * (1.f - wx)),         h11 = (f16)(wy * wx);
      v16h b = v00 * h00;
      b += v01 * h01;
      b += v10 * h10;
      b += v11 * h11;
      c0 = wmma32(afrag(sW, 288, 0, tap * 32), b, c0);
      c1 = wmma32(afrag(sW, 288, 1, tap * 32), b, c1);
    }
    store_nhwc<true>(out, t * 16, 0, c0, bb0);
    store_nhwc<true>(out, t * 16, 1, c1, bb1);
  }
}

// fusion: out[v] = feature[v] + bias + Wf * [cv ; aligned neighbors] (fp32 NCHW out)
__global__ void k_fuse(const f16* __restrict__ feat, const float* __restrict__ featf,
                       const f16* __restrict__ aligned,
                       const f16* __restrict__ Wf3, const f16* __restrict__ Wf5,
                       const f16* __restrict__ Wf8,
                       const float* __restrict__ bf3, const float* __restrict__ bf5,
                       const float* __restrict__ bf8, float* __restrict__ outp) {
  __shared__ f16 sW[32 * 288];
  int v = blockIdx.x;
  int n = NCNT[v], s = STARTS[v];
  const f16* W; const float* bias;
  if (n == 3)      { W = Wf3; bias = bf3; }
  else if (n == 5) { W = Wf5; bias = bf5; }
  else             { W = Wf8; bias = bf8; }
  int Krow = (n + 1) * 32;
  stage_lds(sW, W, 32 * Krow);
  const f16* cv  = feat + (size_t)v * HW * C32;
  const float* cvf = featf + (size_t)v * C32 * HW;
  float* out = outp + (size_t)v * C32 * HW;
  int wave = threadIdx.x >> 5;
  int l = laneid();
  int mo = (l >> 4) << 3;
  v8f bb0 = load_bias(bias, 0), bb1 = load_bias(bias, 1);
  int tbase = blockIdx.y * 64 + wave * 8;
  for (int it = 0; it < 8; it += 2) {
    int p0 = (tbase + it) * 16, p1 = p0 + 16;
    v8f c00 = {}, c01 = {}, c10 = {}, c11 = {};
    {
      v16h b0 = bfrag(cv, p0), b1 = bfrag(cv, p1);
      v16h a0 = afrag(sW, Krow, 0, 0), a1 = afrag(sW, Krow, 1, 0);
      c00 = wmma32(a0, b0, c00); c01 = wmma32(a0, b1, c01);
      c10 = wmma32(a1, b0, c10); c11 = wmma32(a1, b1, c11);
    }
    for (int jn = 0; jn < n; ++jn) {
      const f16* src = aligned + (size_t)(s + jn) * HW * C32;
      v16h b0 = bfrag(src, p0), b1 = bfrag(src, p1);
      int kc = (jn + 1) * 32;
      v16h a0 = afrag(sW, Krow, 0, kc), a1 = afrag(sW, Krow, 1, kc);
      c00 = wmma32(a0, b0, c00); c01 = wmma32(a0, b1, c01);
      c10 = wmma32(a1, b0, c10); c11 = wmma32(a1, b1, c11);
    }
    for (int su = 0; su < 2; ++su) {
      int pix = (su ? p1 : p0) + (l & 15);
      v8f cl = su ? c01 : c00, ch = su ? c11 : c10;
      for (int j = 0; j < 8; ++j) {
        int c = mo + j;
        out[(size_t)c * HW + pix] = cvf[(size_t)c * HW + pix] + bb0[j] + cl[j];
      }
      for (int j = 0; j < 8; ++j) {
        int c = 16 + mo + j;
        out[(size_t)c * HW + pix] = cvf[(size_t)c * HW + pix] + bb1[j] + ch[j];
      }
    }
  }
}

// ---- host side -------------------------------------------------------------
extern "C" void kernel_launch(void* const* d_in, const int* in_sizes, int n_in,
                              void* d_out, int out_size, void* d_ws, size_t ws_size,
                              hipStream_t stream) {
  const float* feature = (const float*)d_in[0];
  const float* w_conv1 = (const float*)d_in[1];
  const float* b_conv1 = (const float*)d_in[2];
  const float* w_a1    = (const float*)d_in[3];
  const float* w_a2    = (const float*)d_in[4];
  const float* w_a3    = (const float*)d_in[5];
  const float* w_at    = (const float*)d_in[6];
  const float* w_off   = (const float*)d_in[7];
  const float* b_off   = (const float*)d_in[8];
  const float* w_dcn   = (const float*)d_in[9];
  const float* b_dcn   = (const float*)d_in[10];
  const float* w_f3    = (const float*)d_in[11];
  const float* b_f3    = (const float*)d_in[12];
  const float* w_f5    = (const float*)d_in[13];
  const float* b_f5    = (const float*)d_in[14];
  const float* w_f8    = (const float*)d_in[15];
  const float* b_f8    = (const float*)d_in[16];
  float* out = (float*)d_out;

  char* ws = (char*)d_ws;
  size_t o = 0;
  auto take = [&](size_t bytes) { char* p = ws + o; o = (o + bytes + 255) & ~(size_t)255; return p; };

  f16* W1   = (f16*)take(2048 * sizeof(f16));           // [32][64]
  f16* Wa   = (f16*)take(3 * 9216 * sizeof(f16));       // 3 x [32][288]
  f16* Wat  = (f16*)take(27648 * sizeof(f16));          // [32][864]
  f16* Woff = (f16*)take(1024 * sizeof(f16));           // [32][32]
  f16* Wd   = (f16*)take(9216 * sizeof(f16));           // [32][288]
  f16* Wf3  = (f16*)take(4096 * sizeof(f16));           // [32][128]
  f16* Wf5  = (f16*)take(6144 * sizeof(f16));           // [32][192]
  f16* Wf8  = (f16*)take(9216 * sizeof(f16));           // [32][288]
  f16* feat16   = (f16*)take((size_t)NVIEW * HW * C32 * sizeof(f16));
  f16* buf16    = (f16*)take((size_t)NPAIR * HW * C32 * sizeof(f16));
  f16* b3buf    = (f16*)take((size_t)NPAIR * 3 * HW * C32 * sizeof(f16));
  float* offs   = (float*)take((size_t)NPAIR * HW * OFFS_STRIDE * sizeof(float));
  f16* aligned  = (f16*)take((size_t)NPAIR * HW * C32 * sizeof(f16));
  (void)ws_size; (void)in_sizes; (void)n_in; (void)out_size;

  auto blocks = [](int n) { return (n + 255) / 256; };

  k_cvt_direct<<<blocks(2048), 256, 0, stream>>>(w_conv1, W1, 2048);
  k_cvt_oihw  <<<blocks(9216), 256, 0, stream>>>(w_a1, Wa + 0 * 9216, 32);
  k_cvt_oihw  <<<blocks(9216), 256, 0, stream>>>(w_a2, Wa + 1 * 9216, 32);
  k_cvt_oihw  <<<blocks(9216), 256, 0, stream>>>(w_a3, Wa + 2 * 9216, 32);
  k_cvt_oihw  <<<blocks(27648), 256, 0, stream>>>(w_at, Wat, 96);
  k_cvt_off   <<<blocks(1024), 256, 0, stream>>>(w_off, Woff);
  k_cvt_oihw  <<<blocks(9216), 256, 0, stream>>>(w_dcn, Wd, 32);
  k_cvt_direct<<<blocks(4096), 256, 0, stream>>>(w_f3, Wf3, 4096);
  k_cvt_direct<<<blocks(6144), 256, 0, stream>>>(w_f5, Wf5, 6144);
  k_cvt_direct<<<blocks(9216), 256, 0, stream>>>(w_f8, Wf8, 9216);
  k_feat_nhwc <<<blocks(NVIEW * HW * C32), 256, 0, stream>>>(feature, feat16);

  dim3 blk(256);
  k_conv1      <<<dim3(NPAIR, 4),     blk, 0, stream>>>(feat16, W1, b_conv1, buf16);
  k_aspp_branch<<<dim3(NPAIR * 3, 4), blk, 0, stream>>>(buf16, Wa, b3buf);
  k_aspp_fuse  <<<dim3(NPAIR, 4),     blk, 0, stream>>>(b3buf, Wat, buf16);
  k_offset     <<<dim3(NPAIR, 4),     blk, 0, stream>>>(buf16, Woff, b_off, offs);
  k_dcn        <<<dim3(NPAIR, 4),     blk, 0, stream>>>(feat16, offs, Wd, b_dcn, aligned);
  k_fuse       <<<dim3(NVIEW, 4),     blk, 0, stream>>>(feat16, feature, aligned,
                                                        Wf3, Wf5, Wf8, b_f3, b_f5, b_f8, out);
}
